// HGTEncoder_15290083574225
// MI455X (gfx1250) — matrix operand order repaired and checked by reading, not compiled, not run on hardware
//
#include <hip/hip_runtime.h>
#include <hip/hip_bf16.h>
#include <math.h>

typedef __attribute__((ext_vector_type(16))) _Float16 v16h;
typedef __attribute__((ext_vector_type(8)))  float    v8f;

#define HIDDEN 128
#define NP 50000
#define NA 20000
#define TPB_TILES 4
#define SWP (128 + 8)   // padded LDS stride (halves) for transposed W

__device__ __forceinline__ float gelu_exact(float x) {
  return 0.5f * x * (1.0f + erff(x * 0.70710678118654752f));
}

// ---------------------------------------------------------------------------
// C[M,128] = op(A[M,K]) @ W[K,128] + bias, optionally:
//   op = exact GELU on A elements (GELU)
//   C = gate*C + (1-gate)*resid, gate = sigmoid(skip[0])   (RESID)
// Compile-time K in {64,128} -> guard-free fully unrolled WMMA chain.
// Persistent block over TPB_TILES m-tiles of 16 rows. 8 waves = 8 N-tiles.
// W staged once per block, transposed+padded in LDS so each lane's B fragment
// is 16 contiguous halves (2x ds_load_b128); B fragments hoisted into VGPRs.
// A fragments preloaded into distinct VGPRs so ds loads pipeline ahead of the
// WMMA chain. WMMA: v_wmma_f32_16x16x32_f16, f32 accumulate. M % 16 == 0.
// ---------------------------------------------------------------------------
template <int K, bool GELU, bool RESID>
__global__ __launch_bounds__(256)
void gemm_wmma_128(const float* __restrict__ A,
                   const float* __restrict__ W,
                   const float* __restrict__ bias,
                   float* __restrict__ C,
                   const float* __restrict__ resid,
                   const float* __restrict__ skip,
                   int numTiles) {
  constexpr int SAP = K + 8;
  constexpr int NT  = K / 32;                // K-tiles
  __shared__ _Float16 sWt[128 * SWP];        // transposed: sWt[n*SWP + k]
  __shared__ _Float16 sA[16 * SAP];          // sA[r*SAP + k]
  const int tid = threadIdx.x;

  // ---- stage W transposed (once per block), float4 coalesced reads ----
  for (int i = tid; i < K * 32; i += 256) {  // K*128/4 float4 elements
    const int k  = i >> 5;                   // W row
    const int n4 = (i & 31) << 2;            // W col base
    const float4 w4 = ((const float4*)W)[i];
    sWt[(n4 + 0) * SWP + k] = (_Float16)w4.x;
    sWt[(n4 + 1) * SWP + k] = (_Float16)w4.y;
    sWt[(n4 + 2) * SWP + k] = (_Float16)w4.z;
    sWt[(n4 + 3) * SWP + k] = (_Float16)w4.w;
  }
  __syncthreads();

  const int wave = tid >> 5, lane = tid & 31;
  const int col0 = wave * 16, lrow = lane & 15, hi = (lane >> 4) & 1;
  const int colg = col0 + lrow;

  // ---- B fragments: loop-invariant per wave, keep in VGPRs ----
  v16h bf[NT];
#pragma unroll
  for (int kt = 0; kt < NT; ++kt) {
    const int bk = kt * 32 + hi * 16;        // lanes 16-31 take K+16
#pragma unroll
    for (int i = 0; i < 16; ++i) bf[kt][i] = sWt[colg * SWP + bk + i];
  }

  const float bv = bias[colg];
  float gate = 0.f, omg = 0.f;
  if (RESID) { gate = 1.f / (1.f + expf(-skip[0])); omg = 1.f - gate; }

  const int t0 = blockIdx.x * TPB_TILES;
  const int t1 = min(t0 + TPB_TILES, numTiles);
  for (int t = t0; t < t1; ++t) {
    const int m0 = t * 16;
    __syncthreads();                         // previous iteration readers done
#pragma unroll
    for (int j = 0; j < K / 64; ++j) {       // 16*K/4/256 float4 per thread
      const int i  = j * 256 + tid;
      const int r  = i / (K / 4);
      const int k4 = (i - r * (K / 4)) << 2;
      float4 a4 = ((const float4*)(A + (size_t)(m0 + r) * K + k4))[0];
      if (GELU) {
        a4.x = gelu_exact(a4.x); a4.y = gelu_exact(a4.y);
        a4.z = gelu_exact(a4.z); a4.w = gelu_exact(a4.w);
      }
      sA[r * SAP + k4 + 0] = (_Float16)a4.x;
      sA[r * SAP + k4 + 1] = (_Float16)a4.y;
      sA[r * SAP + k4 + 2] = (_Float16)a4.z;
      sA[r * SAP + k4 + 3] = (_Float16)a4.w;
      // prefetch next m-tile's A rows into cache (global_prefetch_b8)
      if (t + 1 < t1)
        __builtin_prefetch(A + (size_t)(m0 + 16 + r) * K + k4, 0, 1);
    }
    __syncthreads();

    // preload all A fragments so the ds_load_b128s pipeline ahead of the WMMAs
    v16h afr[NT];
#pragma unroll
    for (int kt = 0; kt < NT; ++kt) {
      const int ak = kt * 32 + hi * 16;      // lanes 16-31 take K+16
#pragma unroll
      for (int i = 0; i < 8; ++i) {
        afr[kt][i]     = sA[lrow * SAP + ak + i];        // K 0..7   (+16 hi)
        afr[kt][i + 8] = sA[lrow * SAP + ak + 8 + i];    // K 8..15  (+16 hi)
      }
    }
    v8f c = {};
#pragma unroll
    for (int kt = 0; kt < NT; ++kt)
      c = __builtin_amdgcn_wmma_f32_16x16x32_f16(false, afr[kt], false, bf[kt],
                                                 (short)0, c, false, false);

    // epilogue: single base pointer, immediate-offset stores (r*512B)
    float* cp = C + (size_t)(m0 + hi * 8) * 128 + colg;
    float rv[8];
    if (RESID) {
      const float* rp = resid + (size_t)(m0 + hi * 8) * 128 + colg;
#pragma unroll
      for (int r = 0; r < 8; ++r) rv[r] = rp[r * 128];
    }
#pragma unroll
    for (int r = 0; r < 8; ++r) {            // C layout: VGPR r -> row r / r+8
      float v = c[r] + bv;
      if (RESID) v = gate * v + omg * rv[r];
      cp[r * 128] = v;
    }
  }
}

// ---------------------------------------------------------------------------
// Fold the per-head relation matrix into the k/v projection:
//   Wc[k, h*32+e] = sum_d Wk[k, h*32+d] * rel[h,d,e]   (row 128 -> bias)
// ---------------------------------------------------------------------------
__global__ void combine_rel(const float* __restrict__ Wk, const float* __restrict__ bk,
                            const float* __restrict__ rel,
                            float* __restrict__ Wc, float* __restrict__ bc) {
  const int idx = blockIdx.x * blockDim.x + threadIdx.x;
  if (idx >= 129 * 128) return;
  const int row = idx >> 7;      // 0..127 weight row, 128 => bias
  const int n   = idx & 127;
  const int h   = n >> 5, e = n & 31;
  float s = 0.f;
  if (row < 128) {
#pragma unroll
    for (int d = 0; d < 32; ++d) s += Wk[row * 128 + h * 32 + d] * rel[h * 1024 + d * 32 + e];
    Wc[row * 128 + n] = s;
  } else {
#pragma unroll
    for (int d = 0; d < 32; ++d) s += bk[h * 32 + d] * rel[h * 1024 + d * 32 + e];
    bc[n] = s;
  }
}

__global__ void fill_kernel(float* __restrict__ p, float v, size_t n) {
  size_t i = (size_t)blockIdx.x * blockDim.x + threadIdx.x;
  const size_t stride = (size_t)gridDim.x * blockDim.x;
  for (; i < n; i += stride) p[i] = v;
}

__device__ __forceinline__ void atomicMaxF(float* addr, float val) {
  if (val >= 0.f) atomicMax((int*)addr, __float_as_int(val));
  else            atomicMin((unsigned int*)addr, (unsigned int)__float_as_int(val));
}

// logits[e,h] = scale * p_rel[h] * dot32(q[di[e],h], krel[si[e],h])
__global__ void edge_logits(const float* __restrict__ q, const float* __restrict__ krel,
                            const int* __restrict__ si, const int* __restrict__ di,
                            const float* __restrict__ prel,
                            float* __restrict__ logits, int EH) {
  const int t = blockIdx.x * blockDim.x + threadIdx.x;
  if (t >= EH) return;
  const int e = t >> 2, h = t & 3;
  const float4* qp = (const float4*)(q    + (size_t)di[e] * 128 + h * 32);
  const float4* kp = (const float4*)(krel + (size_t)si[e] * 128 + h * 32);
  float acc = 0.f;
#pragma unroll
  for (int i = 0; i < 8; ++i) {
    const float4 a = qp[i], b = kp[i];
    acc += a.x * b.x + a.y * b.y + a.z * b.z + a.w * b.w;
  }
  logits[t] = acc * prel[h] * 0.17677669529663687f; // 1/sqrt(32)
}

__global__ void seg_max_k(const float* __restrict__ logits, const int* __restrict__ di,
                          float* __restrict__ m, int EH) {
  const int t = blockIdx.x * blockDim.x + threadIdx.x;
  if (t >= EH) return;
  atomicMaxF(&m[(size_t)di[t >> 2] * 4 + (t & 3)], logits[t]);
}

__global__ void seg_sum_k(const float* __restrict__ logits, const int* __restrict__ di,
                          const float* __restrict__ m, float* __restrict__ s, int EH) {
  const int t = blockIdx.x * blockDim.x + threadIdx.x;
  if (t >= EH) return;
  const int d4 = di[t >> 2] * 4 + (t & 3);
  unsafeAtomicAdd(&s[d4], expf(logits[t] - m[d4]));
}

__global__ void seg_acc_k(const float* __restrict__ logits,
                          const int* __restrict__ si, const int* __restrict__ di,
                          const float* __restrict__ m, const float* __restrict__ s,
                          const float* __restrict__ vrel, float* __restrict__ acc, int EH) {
  const int t = blockIdx.x * blockDim.x + threadIdx.x;
  if (t >= EH) return;
  const int e = t >> 2, h = t & 3;
  const int d = di[e];
  const int d4 = d * 4 + h;
  const float w = expf(logits[t] - m[d4]) / (s[d4] + 1e-16f);
  const float4* vp = (const float4*)(vrel + (size_t)si[e] * 128 + h * 32);
  float* ap = acc + (size_t)d * 128 + h * 32;
#pragma unroll
  for (int i = 0; i < 8; ++i) {
    const float4 v4 = vp[i];
    unsafeAtomicAdd(&ap[4 * i + 0], v4.x * w);
    unsafeAtomicAdd(&ap[4 * i + 1], v4.y * w);
    unsafeAtomicAdd(&ap[4 * i + 2], v4.z * w);
    unsafeAtomicAdd(&ap[4 * i + 3], v4.w * w);
  }
}

// ---------------------------------------------------------------------------
// Host orchestration
// ---------------------------------------------------------------------------
extern "C" void kernel_launch(void* const* d_in, const int* in_sizes, int n_in,
                              void* d_out, int out_size, void* d_ws, size_t ws_size,
                              hipStream_t stream) {
  (void)n_in; (void)out_size; (void)ws_size;
  const float* x_paper  = (const float*)d_in[0];
  const float* x_author = (const float*)d_in[1];
  // params pytree (jax-sorted): layers[0..1] then lin; per-layer leaf offsets:
  //  0 a_lin.author.b  1 a_lin.author.w  2 a_lin.paper.b  3 a_lin.paper.w
  //  4 a_rel.cites 5 a_rel.writes 6 a_rel.written_by
  //  7 k.author.b 8 k.author.w 9 k.paper.b 10 k.paper.w
  // 11 m_rel.cites 12 m_rel.writes 13 m_rel.written_by
  // 14 p_rel.cites 15 p_rel.writes 16 p_rel.written_by
  // 17 q.author.b 18 q.author.w 19 q.paper.b 20 q.paper.w
  // 21 skip.author 22 skip.paper
  // 23 v.author.b 24 v.author.w 25 v.paper.b 26 v.paper.w
  const int LIN = 2 + 27 * 2;             // 56: lin.author.b,w lin.paper.b,w
  const int EIX = LIN + 4;                // 60: edge index arrays
  const int E   = in_sizes[EIX];          // 400000

  auto P  = [&](int i) { return (const float*)d_in[i]; };
  auto IX = [&](int i) { return (const int*)d_in[i]; };

  // workspace carve (fp32)
  float* ws = (float*)d_ws;
  size_t off = 0;
  auto carve = [&](size_t n) { float* p = ws + off; off += n; return p; };
  float* hP0  = carve((size_t)NP * 128);
  float* hA0  = carve((size_t)NA * 128);
  float* hP1  = carve((size_t)NP * 128);
  float* hA1  = carve((size_t)NA * 128);
  float* qP   = carve((size_t)NP * 128);
  float* qA   = carve((size_t)NA * 128);
  float* krel = carve((size_t)NP * 128);
  float* vrel = carve((size_t)NP * 128);
  float* accP = carve((size_t)NP * 128);
  float* accA = carve((size_t)NA * 128);
  float* logits = carve((size_t)E * 4);
  float* mbuf = carve((size_t)NP * 4);
  float* sbuf = carve((size_t)NP * 4);
  float* Wc1 = carve(128 * 128); float* bc1 = carve(128);
  float* Wc2 = carve(128 * 128); float* bc2 = carve(128);

  float* outP = (float*)d_out;
  float* outA = (float*)d_out + (size_t)NP * 128;

  const int EH = E * 4;
  const dim3 B256(256);
  const int gEH  = (EH + 255) / 256;
  const int gCmb = (129 * 128 + 255) / 256;

  auto gemm128 = [&](const float* A, const float* W, const float* b, float* C, int M) {
    const int numTiles = M / 16;
    const int grid = (numTiles + TPB_TILES - 1) / TPB_TILES;
    gemm_wmma_128<128, false, false><<<grid, B256, 0, stream>>>(
        A, W, b, C, nullptr, nullptr, numTiles);
  };
  auto gemm64 = [&](const float* A, const float* W, const float* b, float* C, int M) {
    const int numTiles = M / 16;
    const int grid = (numTiles + TPB_TILES - 1) / TPB_TILES;
    gemm_wmma_128<64, false, false><<<grid, B256, 0, stream>>>(
        A, W, b, C, nullptr, nullptr, numTiles);
  };
  auto gemmOut = [&](const float* A, const float* W, const float* b, float* C, int M,
                     const float* resid, const float* skip) {
    const int numTiles = M / 16;
    const int grid = (numTiles + TPB_TILES - 1) / TPB_TILES;
    gemm_wmma_128<128, true, true><<<grid, B256, 0, stream>>>(
        A, W, b, C, resid, skip, numTiles);
  };

  // -------- input projections --------
  gemm128(x_paper, P(LIN + 3), P(LIN + 2), hP0, NP);
  gemm64(x_author, P(LIN + 1), P(LIN + 0), hA0, NA);

  for (int l = 0; l < 2; ++l) {
    const int Lb = 2 + 27 * l;
    const float* inP = l ? hP1 : hP0;
    const float* inA = l ? hA1 : hA0;
    float* oP = l ? outP : hP1;
    float* oA = l ? outA : hA1;

    // q projections (per dst type)
    gemm128(inP, P(Lb + 20), P(Lb + 19), qP, NP);
    gemm128(inA, P(Lb + 18), P(Lb + 17), qA, NA);

    fill_kernel<<<2048, B256, 0, stream>>>(accP, 0.f, (size_t)NP * 128);
    fill_kernel<<<1024, B256, 0, stream>>>(accA, 0.f, (size_t)NA * 128);

    // edge type: src feats / Msrc / kv weight idx / rel idx / q,acc of dst / Ndst / si,di
    struct ESpec { const float* xs; int Ms; int kb, kw, vb, vw, ar, mr, pr;
                   const float* qd; float* accd; int Nd; int si, di; };
    const ESpec specs[3] = {
      { inP, NP, Lb + 9, Lb + 10, Lb + 25, Lb + 26, Lb + 4, Lb + 11, Lb + 14, qP, accP, NP, EIX + 0, EIX + 1 }, // cites  p->p
      { inA, NA, Lb + 7, Lb + 8,  Lb + 23, Lb + 24, Lb + 5, Lb + 12, Lb + 15, qP, accP, NP, EIX + 2, EIX + 3 }, // writes a->p
      { inP, NP, Lb + 9, Lb + 10, Lb + 25, Lb + 26, Lb + 6, Lb + 13, Lb + 16, qA, accA, NA, EIX + 4, EIX + 5 }, // written_by p->a
    };

    for (int es = 0; es < 3; ++es) {
      const ESpec& S = specs[es];
      // fold relation matrices into projection weights, then one GEMM each
      combine_rel<<<gCmb, B256, 0, stream>>>(P(S.kw), P(S.kb), P(S.ar), Wc1, bc1);
      combine_rel<<<gCmb, B256, 0, stream>>>(P(S.vw), P(S.vb), P(S.mr), Wc2, bc2);
      gemm128(S.xs, Wc1, bc1, krel, S.Ms);
      gemm128(S.xs, Wc2, bc2, vrel, S.Ms);

      edge_logits<<<gEH, B256, 0, stream>>>(S.qd, krel, IX(S.si), IX(S.di), P(S.pr), logits, EH);
      fill_kernel<<<64, B256, 0, stream>>>(mbuf, -INFINITY, (size_t)S.Nd * 4);
      fill_kernel<<<64, B256, 0, stream>>>(sbuf, 0.f, (size_t)S.Nd * 4);
      seg_max_k<<<gEH, B256, 0, stream>>>(logits, IX(S.di), mbuf, EH);
      seg_sum_k<<<gEH, B256, 0, stream>>>(logits, IX(S.di), mbuf, sbuf, EH);
      seg_acc_k<<<gEH, B256, 0, stream>>>(logits, IX(S.si), IX(S.di), mbuf, sbuf, vrel, S.accd, EH);
    }

    // out = sigmoid(skip)*(gelu(acc) @ a_lin + b) + (1-sigmoid(skip))*x_in
    gemmOut(accP, P(Lb + 3), P(Lb + 2), oP, NP, inP, P(Lb + 22));
    gemmOut(accA, P(Lb + 1), P(Lb + 0), oA, NA, inA, P(Lb + 21));
  }
}
